// Encoder_20349555049068
// MI455X (gfx1250) — matrix operand order repaired and checked by reading, not compiled, not run on hardware
//
#include <hip/hip_runtime.h>

// ---------------- problem constants ----------------
#define MODEL_DIM   512
#define HEAD_DIM    64
#define QKV_DIM     192                 // q|k|v concatenated
#define FORWARD_DIM 2048
#define SEQ         2048
#define BATCH       8
#define NTOK        (BATCH * SEQ)       // 16384 rows
#define N_LAYERS    6

typedef __attribute__((ext_vector_type(16))) _Float16 v16h;
typedef __attribute__((ext_vector_type(8)))  float    v8f;
typedef _Float16 half_t;

static __device__ __forceinline__ v8f wmma_f16(v16h a, v16h b, v8f c) {
  // D = A(16x32 f16) * B(32x16 f16) + C(16x16 f32)
  return __builtin_amdgcn_wmma_f32_16x16x32_f16(
      /*neg_a=*/false, a, /*neg_b=*/false, b,
      /*c_mod=*/(short)0, c, /*reuse_a=*/false, /*reuse_b=*/false);
}

// Swizzled B layout: element (kk,col) of a [K,N] B matrix lives at
//   [kt = kk/32][ct = col/16][lane = (kk%32)/16*16 + col%16][e = kk%16]
// so each lane's 16-element WMMA B fragment is one contiguous 32-byte run.
static __device__ __forceinline__ size_t bsw_off(int kk, int col, int N) {
  int kt = kk >> 5, kl = kk & 31;
  int hi = kl >> 4, e  = kl & 15;
  int ct = col >> 4, lo = col & 15;
  int lane = hi * 16 + lo;
  return ((((size_t)kt * (N >> 4) + ct) * 32) + lane) * 16 + e;
}

// ---------------- positional encoding (reproduces reference's seq-axis bug) ---
__global__ void pos_enc_kernel(const float* __restrict__ X,
                               float* __restrict__ out,
                               half_t* __restrict__ outh, int total) {
  int i = blockIdx.x * blockDim.x + threadIdx.x;
  if (i >= total) return;
  int d = i % MODEL_DIM;
  int s = (i / MODEL_DIM) % SEQ;
  float f  = __powf(1e-4f, (float)d / (float)MODEL_DIM);
  float p  = (float)s * f;
  float pe = ((s & 1) == 0) ? __cosf(p) : __sinf(p);
  float v  = X[i] + pe;
  out[i]  = v;
  outh[i] = (half_t)v;
}

// ---------------- one-time weight prep (weights shared by all 6 layers) -------
// concat wq|wk|wv -> f16 B-swizzled [512,192]; bq|bk|bv -> f32 [192]
__global__ void build_wqkv_kernel(const float* __restrict__ wq, const float* __restrict__ bq,
                                  const float* __restrict__ wk, const float* __restrict__ bk,
                                  const float* __restrict__ wv, const float* __restrict__ bv,
                                  half_t* __restrict__ wqkv, float* __restrict__ bqkv) {
  int i = blockIdx.x * blockDim.x + threadIdx.x;
  if (i >= MODEL_DIM * QKV_DIM) return;
  int r = i / QKV_DIM, c = i % QKV_DIM;
  int sel = c / HEAD_DIM, cc = c % HEAD_DIM;
  const float* w = (sel == 0) ? wq : (sel == 1) ? wk : wv;
  wqkv[bsw_off(r, c, QKV_DIM)] = (half_t)w[(size_t)r * HEAD_DIM + cc];
  if (i < QKV_DIM) {
    int s2 = i / HEAD_DIM, c2 = i % HEAD_DIM;
    const float* b = (s2 == 0) ? bq : (s2 == 1) ? bk : bv;
    bqkv[i] = b[c2];
  }
}

// wo_eff[d,n] = sum_h wo[h*64+d, n] (all heads identical) -> f16 B-swizzled
__global__ void wo_eff_kernel(const float* __restrict__ wo, half_t* __restrict__ woe) {
  int i = blockIdx.x * blockDim.x + threadIdx.x;
  if (i >= HEAD_DIM * MODEL_DIM) return;
  int d = i / MODEL_DIM;
  int n = i % MODEL_DIM;
  float s = 0.f;
#pragma unroll
  for (int h = 0; h < 8; ++h)
    s += wo[(size_t)(h * HEAD_DIM + d) * MODEL_DIM + n];
  woe[bsw_off(d, n, MODEL_DIM)] = (half_t)s;
}

// f32 [K,N] -> f16 B-swizzled
__global__ void cvt_bsw_kernel(const float* __restrict__ src, half_t* __restrict__ dst,
                               int K, int N) {
  int i = blockIdx.x * blockDim.x + threadIdx.x;
  if (i >= K * N) return;
  int r = i / N, c = i % N;
  dst[bsw_off(r, c, N)] = (half_t)src[i];
}

// V rows of the fused qkv buffer -> per-batch B-swizzled layout for flash PV
__global__ void swizzle_v_kernel(const half_t* __restrict__ qkv, half_t* __restrict__ vsw) {
  int i = blockIdx.x * blockDim.x + threadIdx.x;
  if (i >= NTOK * HEAD_DIM) return;
  int row = i / HEAD_DIM;   // global token
  int dim = i % HEAD_DIM;
  int b = row / SEQ, s = row % SEQ;
  half_t val = qkv[(size_t)row * QKV_DIM + 2 * HEAD_DIM + dim];
  vsw[(size_t)b * SEQ * HEAD_DIM + bsw_off(s, dim, HEAD_DIM)] = val;
}

// ---------------- WMMA GEMM: C = op(A[MxK]*Bsw[KxN] + bias), all f16 ----------
// block = 128 threads = 4 waves; each wave -> 32x32 tile; workgroup -> 64x64
__global__ void __launch_bounds__(128) wmma_gemm_kernel(
    const half_t* __restrict__ A, const half_t* __restrict__ Bsw,
    const float* __restrict__ bias, half_t* __restrict__ C,
    int M, int N, int K, int relu) {
  int wave = threadIdx.x >> 5;
  int lane = threadIdx.x & 31;
  int lo = lane & 15, hi = lane >> 4;
  int mbase = blockIdx.y * 64 + (wave >> 1) * 32;
  int nbase = blockIdx.x * 64 + (wave & 1) * 32;
  int ntiles = N >> 4;
  const v16h* Bv = (const v16h*)Bsw;

  v8f acc[2][2];
#pragma unroll
  for (int i = 0; i < 2; ++i)
#pragma unroll
    for (int j = 0; j < 2; ++j)
#pragma unroll
      for (int e = 0; e < 8; ++e) acc[i][j][e] = 0.f;

  for (int k0 = 0; k0 < K; k0 += 32) {
    v16h a[2], b[2];
#pragma unroll
    for (int rt = 0; rt < 2; ++rt) {
      const half_t* arow = A + (size_t)(mbase + rt * 16 + lo) * K + k0;
      __builtin_prefetch(arow + 32, 0, 0);  // next K slice -> global_prefetch_b8
#pragma unroll
      for (int e = 0; e < 16; ++e) {
        int kk = hi * 16 + e;
        a[rt][e] = arow[kk];                // one contiguous 32B run
      }
    }
#pragma unroll
    for (int ct = 0; ct < 2; ++ct) {
      int ctg = (nbase >> 4) + ct;
      b[ct] = Bv[((size_t)(k0 >> 5) * ntiles + ctg) * 32 + lane];  // 32B vector load
    }
#pragma unroll
    for (int i = 0; i < 2; ++i)
#pragma unroll
      for (int j = 0; j < 2; ++j)
        acc[i][j] = wmma_f16(a[i], b[j], acc[i][j]);
  }

#pragma unroll
  for (int i = 0; i < 2; ++i)
#pragma unroll
    for (int j = 0; j < 2; ++j) {
      int col = nbase + j * 16 + lo;
      float bb = bias ? bias[col] : 0.f;
#pragma unroll
      for (int r = 0; r < 8; ++r) {
        int row = mbase + i * 16 + r + 8 * hi;
        float v0 = acc[i][j][r] + bb;
        if (relu) v0 = fmaxf(v0, 0.f);
        C[(size_t)row * N + col] = (half_t)v0;
      }
    }
}

// ---------------- flash attention (f16 q/k from fused qkv, swizzled V) --------
// block = 128 threads = 4 waves; each wave owns 32 query rows; keys in chunks of 32
__global__ void __launch_bounds__(128) attention_kernel(
    const half_t* __restrict__ qkv, const half_t* __restrict__ vsw,
    half_t* __restrict__ att) {
  const int HD = HEAD_DIM;
  const int RS = QKV_DIM;  // row stride in qkv buffer
  int wave = threadIdx.x >> 5;
  int lane = threadIdx.x & 31;
  int lo = lane & 15, hi = lane >> 4;
  int b = blockIdx.y;
  int qbase = (blockIdx.x * 4 + wave) * 32;
  const half_t* base = qkv + (size_t)b * SEQ * RS;
  const half_t* qb = base;             // cols [0,64)
  const half_t* kb = base + HEAD_DIM;  // cols [64,128)
  const v16h* vswb = (const v16h*)(vsw + (size_t)b * SEQ * HD);

  __shared__ half_t Pt[4][32][33];  // per-wave prob tile, padded stride

  // Q held in registers as WMMA A-tiles: [rowtile][kstep(64=2x32)]
  v16h aq[2][2];
#pragma unroll
  for (int rt = 0; rt < 2; ++rt)
#pragma unroll
    for (int ks = 0; ks < 2; ++ks) {
      const half_t* qrow = qb + (size_t)(qbase + rt * 16 + lo) * RS + ks * 32;
#pragma unroll
      for (int e = 0; e < 16; ++e) {
        int kk = hi * 16 + e;
        aq[rt][ks][e] = qrow[kk];
      }
    }

  float m[2][8], l[2][8];
  v8f o[2][4];
#pragma unroll
  for (int rt = 0; rt < 2; ++rt)
#pragma unroll
    for (int r = 0; r < 8; ++r) { m[rt][r] = -1e30f; l[rt][r] = 0.f; }
#pragma unroll
  for (int rt = 0; rt < 2; ++rt)
#pragma unroll
    for (int dt = 0; dt < 4; ++dt)
#pragma unroll
      for (int e = 0; e < 8; ++e) o[rt][dt][e] = 0.f;

  const float scale = 0.125f;  // 1/sqrt(64)

  for (int kc = 0; kc < SEQ; kc += 32) {
    // ---- scores S[32q x 32k] = Q * K^T ----
    v8f sc[2][2];
#pragma unroll
    for (int i = 0; i < 2; ++i)
#pragma unroll
      for (int j = 0; j < 2; ++j)
#pragma unroll
        for (int e = 0; e < 8; ++e) sc[i][j][e] = 0.f;

#pragma unroll
    for (int ks = 0; ks < 2; ++ks) {
      v16h bk[2];  // B-tile of K^T: (kdim, key); kd contiguous per lane
#pragma unroll
      for (int nt = 0; nt < 2; ++nt) {
        const half_t* krow = kb + (size_t)(kc + nt * 16 + lo) * RS + ks * 32;
#pragma unroll
        for (int e = 0; e < 16; ++e)
          bk[nt][e] = krow[hi * 16 + e];
      }
#pragma unroll
      for (int rt = 0; rt < 2; ++rt)
#pragma unroll
        for (int nt = 0; nt < 2; ++nt)
          sc[rt][nt] = wmma_f16(aq[rt][ks], bk[nt], sc[rt][nt]);
    }

    // ---- online softmax update (row = r + 8*hi within rowtile) ----
#pragma unroll
    for (int rt = 0; rt < 2; ++rt) {
      float mx[8], mnew[8], alpha[8], psum[8];
#pragma unroll
      for (int r = 0; r < 8; ++r)
        mx[r] = fmaxf(sc[rt][0][r], sc[rt][1][r]) * scale;
#pragma unroll
      for (int sh = 1; sh < 16; sh <<= 1)
#pragma unroll
        for (int r = 0; r < 8; ++r)
          mx[r] = fmaxf(mx[r], __shfl_xor(mx[r], sh, 32));
#pragma unroll
      for (int r = 0; r < 8; ++r) {
        mnew[r]  = fmaxf(m[rt][r], mx[r]);
        alpha[r] = __expf(m[rt][r] - mnew[r]);
        m[rt][r] = mnew[r];
        psum[r]  = 0.f;
      }
#pragma unroll
      for (int nt = 0; nt < 2; ++nt)
#pragma unroll
        for (int r = 0; r < 8; ++r) {
          float p = __expf(sc[rt][nt][r] * scale - mnew[r]);
          psum[r] += p;
          Pt[wave][rt * 16 + r + 8 * hi][nt * 16 + lo] = (half_t)p;
        }
#pragma unroll
      for (int sh = 1; sh < 16; sh <<= 1)
#pragma unroll
        for (int r = 0; r < 8; ++r)
          psum[r] += __shfl_xor(psum[r], sh, 32);
#pragma unroll
      for (int r = 0; r < 8; ++r)
        l[rt][r] = l[rt][r] * alpha[r] + psum[r];
#pragma unroll
      for (int dt = 0; dt < 4; ++dt)
#pragma unroll
        for (int r = 0; r < 8; ++r)
          o[rt][dt][r] *= alpha[r];
    }

    // ---- O += P * V  (P from LDS in A-layout; V as pre-swizzled vector loads)
    v16h ap[2];
#pragma unroll
    for (int rt = 0; rt < 2; ++rt)
#pragma unroll
      for (int e = 0; e < 16; ++e) {
        int kk = hi * 16 + e;
        ap[rt][e] = Pt[wave][rt * 16 + lo][kk];
      }
#pragma unroll
    for (int dt = 0; dt < 4; ++dt) {
      v16h bv = vswb[((size_t)(kc >> 5) * 4 + dt) * 32 + lane];  // 32B vector load
#pragma unroll
      for (int rt = 0; rt < 2; ++rt)
        o[rt][dt] = wmma_f16(ap[rt], bv, o[rt][dt]);
    }
  }

  // ---- normalize and store att[b, row, dim] as f16 ----
  half_t* ab = att + (size_t)b * SEQ * HD;
#pragma unroll
  for (int rt = 0; rt < 2; ++rt)
#pragma unroll
    for (int dt = 0; dt < 4; ++dt) {
      int dim = dt * 16 + lo;
#pragma unroll
      for (int r = 0; r < 8; ++r) {
        int row = qbase + rt * 16 + r + 8 * hi;
        ab[(size_t)row * HD + dim] = (half_t)(o[rt][dt][r] / l[rt][r]);
      }
    }
}

// ---------------- fused residual + LayerNorm over D=512 -----------------------
__global__ void __launch_bounds__(128) residual_ln_kernel(
    const float* __restrict__ x, const half_t* __restrict__ y,
    const float* __restrict__ g, const float* __restrict__ b,
    float* __restrict__ out, half_t* __restrict__ outh) {
  const int D = MODEL_DIM;
  int row = blockIdx.x;
  int t = threadIdx.x;

  float vals[4];
  float s = 0.f;
#pragma unroll
  for (int i = 0; i < 4; ++i) {
    int d = t + i * 128;
    vals[i] = x[(size_t)row * D + d] + (float)y[(size_t)row * D + d];
    s += vals[i];
  }
  __shared__ float red[128];
  red[t] = s; __syncthreads();
  for (int off = 64; off > 0; off >>= 1) {
    if (t < off) red[t] += red[t + off];
    __syncthreads();
  }
  float mean = red[0] * (1.f / D);
  __syncthreads();
  float var = 0.f;
#pragma unroll
  for (int i = 0; i < 4; ++i) { float d0 = vals[i] - mean; var += d0 * d0; }
  red[t] = var; __syncthreads();
  for (int off = 64; off > 0; off >>= 1) {
    if (t < off) red[t] += red[t + off];
    __syncthreads();
  }
  float inv = rsqrtf(red[0] * (1.f / D) + 1e-5f);
#pragma unroll
  for (int i = 0; i < 4; ++i) {
    int d = t + i * 128;
    float v = (vals[i] - mean) * inv * g[d] + b[d];
    out[(size_t)row * D + d] = v;
    if (outh) outh[(size_t)row * D + d] = (half_t)v;
  }
}

// ---------------- host orchestration ------------------------------------------
extern "C" void kernel_launch(void* const* d_in, const int* in_sizes, int n_in,
                              void* d_out, int out_size, void* d_ws, size_t ws_size,
                              hipStream_t stream) {
  (void)in_sizes; (void)n_in; (void)out_size; (void)ws_size;
  const float* X    = (const float*)d_in[0];
  const float* wq   = (const float*)d_in[1];
  const float* bq   = (const float*)d_in[2];
  const float* wk   = (const float*)d_in[3];
  const float* bk   = (const float*)d_in[4];
  const float* wv   = (const float*)d_in[5];
  const float* bv   = (const float*)d_in[6];
  const float* wo   = (const float*)d_in[7];
  const float* bo   = (const float*)d_in[8];
  const float* w1   = (const float*)d_in[9];
  const float* b1   = (const float*)d_in[10];
  const float* w2   = (const float*)d_in[11];
  const float* b2   = (const float*)d_in[12];
  const float* ln1g = (const float*)d_in[13];
  const float* ln1b = (const float*)d_in[14];
  const float* ln2g = (const float*)d_in[15];
  const float* ln2b = (const float*)d_in[16];
  float* out = (float*)d_out;

  // ---- workspace layout ----
  char* ws = (char*)d_ws;
  size_t off = 0;
  auto alloc = [&](size_t bytes) { char* p = ws + off; off += (bytes + 255) & ~(size_t)255; return p; };

  float*  bufA  = (float*) alloc((size_t)NTOK * MODEL_DIM * 4);    // residual f32
  float*  bufB  = (float*) alloc((size_t)NTOK * MODEL_DIM * 4);    // post-LN1 f32
  half_t* bufAh = (half_t*)alloc((size_t)NTOK * MODEL_DIM * 2);    // f16 shadows
  half_t* bufBh = (half_t*)alloc((size_t)NTOK * MODEL_DIM * 2);
  half_t* projh = (half_t*)alloc((size_t)NTOK * MODEL_DIM * 2);    // mh / ff2 out
  half_t* qkvh  = (half_t*)alloc((size_t)NTOK * QKV_DIM * 2);      // fused q|k|v
  half_t* vswh  = (half_t*)alloc((size_t)NTOK * HEAD_DIM * 2);     // swizzled V
  half_t* atth  = (half_t*)alloc((size_t)NTOK * HEAD_DIM * 2);
  half_t* Hh    = (half_t*)alloc((size_t)NTOK * FORWARD_DIM * 2);  // FFN hidden
  half_t* wqkvh = (half_t*)alloc((size_t)MODEL_DIM * QKV_DIM * 2);
  float*  bqkv  = (float*) alloc((size_t)QKV_DIM * 4);
  half_t* woeh  = (half_t*)alloc((size_t)HEAD_DIM * MODEL_DIM * 2);
  half_t* w1h   = (half_t*)alloc((size_t)MODEL_DIM * FORWARD_DIM * 2);
  half_t* w2h   = (half_t*)alloc((size_t)FORWARD_DIM * MODEL_DIM * 2);

  // ---- one-time prep (weights are shared across all 6 layers) ----
  int total = NTOK * MODEL_DIM;
  pos_enc_kernel<<<(total + 255) / 256, 256, 0, stream>>>(X, bufA, bufAh, total);
  build_wqkv_kernel<<<(MODEL_DIM * QKV_DIM + 255) / 256, 256, 0, stream>>>(
      wq, bq, wk, bk, wv, bv, wqkvh, bqkv);
  wo_eff_kernel<<<(HEAD_DIM * MODEL_DIM + 255) / 256, 256, 0, stream>>>(wo, woeh);
  cvt_bsw_kernel<<<(MODEL_DIM * FORWARD_DIM + 255) / 256, 256, 0, stream>>>(
      w1, w1h, MODEL_DIM, FORWARD_DIM);
  cvt_bsw_kernel<<<(FORWARD_DIM * MODEL_DIM + 255) / 256, 256, 0, stream>>>(
      w2, w2h, FORWARD_DIM, MODEL_DIM);

  for (int layer = 0; layer < N_LAYERS; ++layer) {
    // fused QKV: [16384,512] x [512,192]
    wmma_gemm_kernel<<<dim3(QKV_DIM / 64, NTOK / 64), 128, 0, stream>>>(
        bufAh, wqkvh, bqkv, qkvh, NTOK, QKV_DIM, MODEL_DIM, 0);
    // V -> WMMA-B layout for the flash PV loop
    swizzle_v_kernel<<<(NTOK * HEAD_DIM + 255) / 256, 256, 0, stream>>>(qkvh, vswh);
    // flash attention
    attention_kernel<<<dim3(SEQ / 128, BATCH), 128, 0, stream>>>(qkvh, vswh, atth);
    // output projection via head-summed wo_eff: [16384,64] x [64,512]
    wmma_gemm_kernel<<<dim3(MODEL_DIM / 64, NTOK / 64), 128, 0, stream>>>(
        atth, woeh, bo, projh, NTOK, MODEL_DIM, HEAD_DIM, 0);
    residual_ln_kernel<<<NTOK, 128, 0, stream>>>(bufA, projh, ln1g, ln1b, bufB, bufBh);
    // FFN
    wmma_gemm_kernel<<<dim3(FORWARD_DIM / 64, NTOK / 64), 128, 0, stream>>>(
        bufBh, w1h, b1, Hh, NTOK, FORWARD_DIM, MODEL_DIM, 1);
    wmma_gemm_kernel<<<dim3(MODEL_DIM / 64, NTOK / 64), 128, 0, stream>>>(
        Hh, w2h, b2, projh, NTOK, MODEL_DIM, FORWARD_DIM, 0);
    if (layer == N_LAYERS - 1) {
      residual_ln_kernel<<<NTOK, 128, 0, stream>>>(bufB, projh, ln2g, ln2b, out, (half_t*)nullptr);
    } else {
      residual_ln_kernel<<<NTOK, 128, 0, stream>>>(bufB, projh, ln2g, ln2b, bufA, bufAh);
    }
  }
}
